// Event_Coreference_28243704939269
// MI455X (gfx1250) — compile-verified, matrix operand order checked by reading, not run on hardware
//
#include <hip/hip_runtime.h>
#include <hip/hip_bf16.h>
#include <stdint.h>

// ---------------- problem constants ----------------
#define S_LEN   4096
#define D_MODEL 768
#define NHEAD   12
#define DHEAD   64
#define NLAYER  12
#define FFDIM   3072
#define WWIN    256
#define HID     384
#define H4      1536      // 4*HID
#define NEV     64
#define NPAIR   2016

typedef __attribute__((ext_vector_type(16))) __bf16 v16bf;
typedef __attribute__((ext_vector_type(8)))  float  v8f;
typedef __attribute__((ext_vector_type(4)))  unsigned int u32x4;
typedef __attribute__((ext_vector_type(4)))  int i32x4;
typedef __attribute__((ext_vector_type(8)))  int i32x8;

#if __has_builtin(__builtin_amdgcn_tensor_load_to_lds)
#define HAVE_TDM 1
#else
#define HAVE_TDM 0
#endif

// ---------------- CDNA5 primitives ----------------
__device__ __forceinline__ v8f wmma_bf16(v16bf a, v16bf b, v8f c) {
  return __builtin_amdgcn_wmma_f32_16x16x32_bf16(false, a, false, b, (short)0, c,
                                                 false, false);
}

// async global -> LDS, 16 bytes per lane (ASYNCcnt)
__device__ __forceinline__ void async_ld_b128(void* lds_dst, const void* gsrc) {
  unsigned ldsa = (unsigned)(uintptr_t)lds_dst;   // low 32 bits of LDS aperture = LDS offset
  uint64_t ga = (uint64_t)(uintptr_t)gsrc;
  asm volatile("global_load_async_to_lds_b128 %0, %1, off"
               :: "v"(ldsa), "v"(ga) : "memory");
}

__device__ __forceinline__ void wait_async0() {
  asm volatile("s_wait_asynccnt 0x0" ::: "memory");
}

__device__ __forceinline__ void wait_tensor0() {
#if __has_builtin(__builtin_amdgcn_s_wait_tensorcnt)
  __builtin_amdgcn_s_wait_tensorcnt(0);
#else
  asm volatile("s_wait_tensorcnt 0x0" ::: "memory");
#endif
}

#if HAVE_TDM
// TDM: load a 2D tile (rows x cols, 2-byte elements) from a row-major tensor
// (row stride = stride_elems) into LDS at lds_addr. D# per CDNA5 ISA ch.8.
__device__ __forceinline__ void tdm_load_2d(unsigned lds_addr, const void* gptr,
                                            int rows, int cols, int stride_elems,
                                            int tensor_rows) {
  uint64_t ga = (uint64_t)(uintptr_t)gptr;
  u32x4 g0;
  g0[0] = 1u;                                            // count=1, user mode
  g0[1] = lds_addr;                                      // lds_addr (bytes)
  g0[2] = (unsigned)(ga & 0xFFFFFFFFu);                  // global_addr[31:0]
  g0[3] = (unsigned)((ga >> 32) & 0x1FFFFFFu) | (2u << 30); // addr[56:32] | type=2
  unsigned td0 = (unsigned)stride_elems;                 // tensor_dim0 (>= cols)
  unsigned td1 = (unsigned)tensor_rows;                  // tensor_dim1
  i32x8 g1;
  g1[0] = (int)(1u << 16);                               // data_size=2B, no multicast
  g1[1] = (int)((td0 & 0xFFFFu) << 16);                  // [63:48]=tensor_dim0 lo16
  g1[2] = (int)(((td0 >> 16) & 0xFFFFu) | ((td1 & 0xFFFFu) << 16));
  g1[3] = (int)(((td1 >> 16) & 0xFFFFu) | (((unsigned)cols & 0xFFFFu) << 16)); // tile_dim0
  g1[4] = (int)((unsigned)rows & 0xFFFFu);               // tile_dim1 (tile_dim2=0)
  g1[5] = (int)((unsigned)stride_elems);                 // tensor_dim0_stride lo32
  g1[6] = 0;                                             // stride hi16, dim1_stride lo16
  g1[7] = 0;
  i32x4 g2 = {0, 0, 0, 0};
  i32x4 g3 = {0, 0, 0, 0};
#if __clang_major__ >= 23
  i32x8 g4 = {0, 0, 0, 0, 0, 0, 0, 0};
  __builtin_amdgcn_tensor_load_to_lds(g0, g1, g2, g3, g4, 0);
#else
  __builtin_amdgcn_tensor_load_to_lds(g0, g1, g2, g3, 0);
#endif
}
#endif

// Load a 16-bit A/B fragment row pattern (ISA 7.12.2, 16-bit A 16x32):
// lane holds row (lane&15); elem e<8 -> k = k0 + 8*hi + e ; e>=8 -> +16.
__device__ __forceinline__ v16bf fragA(const __bf16* rowp, int lane, int k0) {
  int kb = k0 + ((lane & 16) ? 8 : 0);
  union { uint4 u[2]; v16bf v; } r;
  r.u[0] = *(const uint4*)(rowp + kb);
  r.u[1] = *(const uint4*)(rowp + kb + 16);
  return r.v;
}

__device__ __forceinline__ float gelu_tanh(float x) {
  float x3 = x * x * x;
  return 0.5f * x * (1.0f + tanhf(0.7978845608028654f * (x + 0.044715f * x3)));
}
__device__ __forceinline__ float sigmoidf(float x) {
  return 1.0f / (1.0f + __expf(-x));
}

// ---------------- generic bf16 WMMA GEMM (async/TDM double-buffered) ----------------
// C(MxN) = A(MxK,bf16 row-major) * Bt^T, Bt is (N x K) bf16 row-major.
// grid: ((M+255)/256, N/64), block 256 (8 waves, each: 32 rows x 64 cols).
__global__ void gemm_bf16_kernel(const __bf16* __restrict__ A,
                                 const __bf16* __restrict__ Bt,
                                 const float* __restrict__ bias,
                                 float* __restrict__ C,
                                 __bf16* __restrict__ Cb,
                                 int M, int N, int K, int act) {
  __shared__ __align__(16) __bf16 As[2][256][32];
  __shared__ __align__(16) __bf16 Bs[2][64][32];
  const int bm = blockIdx.x * 256, bn = blockIdx.y * 64;
  const int wid = threadIdx.x >> 5, lane = threadIdx.x & 31;
  const int lr = lane & 15;
  const int hi = (lane & 16) ? 1 : 0;
  v8f acc[2][4] = {};
  const int nk = K / 32;

  auto stage = [&](int kt, int buf) {
    const int k0 = kt * 32;
    // A tile 256x32: one row per thread, 4 async b128 (OOB rows left stale; their
    // outputs are discarded in the epilogue).
    int gr = bm + threadIdx.x;
    if (gr < M) {
      const __bf16* src = A + (size_t)gr * K + k0;
      __bf16* dst = &As[buf][threadIdx.x][0];
#pragma unroll
      for (int c = 0; c < 32; c += 8) async_ld_b128(dst + c, src + c);
    }
    // B tile 64x32 via Tensor Data Mover (wave 0 issues one descriptor).
#if HAVE_TDM
    if (wid == 0)
      tdm_load_2d((unsigned)(uintptr_t)&Bs[buf][0][0],
                  Bt + (size_t)bn * K + k0, 64, 32, K, N);
#else
    {
      int r = threadIdx.x >> 2, c = (threadIdx.x & 3) * 8;
      async_ld_b128(&Bs[buf][r][c], Bt + (size_t)(bn + r) * K + k0 + c);
    }
#endif
  };
  auto drain = [&]() {
    wait_async0();
#if HAVE_TDM
    wait_tensor0();
#endif
    __syncthreads();
  };

  stage(0, 0);
  drain();
  for (int kt = 0; kt < nk; ++kt) {
    int buf = kt & 1;
    if (kt + 1 < nk) stage(kt + 1, (kt + 1) & 1);   // overlap DMA with WMMA
    v16bf a0 = fragA(&As[buf][wid * 32 + lr][0], lane, 0);
    v16bf a1 = fragA(&As[buf][wid * 32 + 16 + lr][0], lane, 0);
#pragma unroll
    for (int t = 0; t < 4; ++t) {
      v16bf bfr = fragA(&Bs[buf][t * 16 + lr][0], lane, 0);
      acc[0][t] = wmma_bf16(a0, bfr, acc[0][t]);
      acc[1][t] = wmma_bf16(a1, bfr, acc[1][t]);
    }
    drain();
  }

  // epilogue: elem j of acc[r2][t] -> (m = wid*32 + r2*16 + j + 8*hi, n = t*16 + lr)
#pragma unroll
  for (int r2 = 0; r2 < 2; ++r2)
#pragma unroll
    for (int t = 0; t < 4; ++t)
#pragma unroll
      for (int j = 0; j < 8; ++j) {
        int m = bm + wid * 32 + r2 * 16 + j + hi * 8;
        int n = bn + t * 16 + lr;
        if (m < M) {
          float val = acc[r2][t][j] + (bias ? bias[n] : 0.0f);
          if (act == 1) val = gelu_tanh(val);
          else if (act == 2) val = fmaxf(val, 0.0f);
          C[(size_t)m * N + n] = val;
          if (Cb) Cb[(size_t)m * N + n] = (__bf16)val;
        }
      }
}

// ---------------- sliding-window flash attention ----------------
// grid (NHEAD, S/W, 2); block 256 (8 waves x 16 query rows = 128 queries).
__global__ void attn_kernel(const __bf16* __restrict__ qb,
                            const __bf16* __restrict__ kb,
                            const __bf16* __restrict__ vb,
                            float* __restrict__ ctx,
                            __bf16* __restrict__ ctxb) {
  const int h = blockIdx.x, c = blockIdx.y, half = blockIdx.z;
  const int wid = threadIdx.x >> 5, lane = threadIdx.x & 31;
  const int lr = lane & 15;
  const int hi = (lane & 16) ? 1 : 0;
  const int qrow0 = c * (2 * 128) + half * 128 + wid * 16;

  __shared__ __align__(16) __bf16 Ks[32][64];
  __shared__ __align__(16) __bf16 Vt[64][32];
  __shared__ float Pl[8][16][32];

  const __bf16* qrowp = qb + (size_t)(qrow0 + lr) * D_MODEL + h * DHEAD;
  v16bf qf0 = fragA(qrowp, lane, 0);
  v16bf qf1 = fragA(qrowp, lane, 32);

  float mrow[8], lrow[8];
#pragma unroll
  for (int j = 0; j < 8; ++j) { mrow[j] = -INFINITY; lrow[j] = 0.0f; }
  v8f ctxa[4] = {};

  for (int it = 0; it < 24; ++it) {            // 24 * 32 = 768 window keys
    int kbase = c * WWIN - WWIN + it * 32;
    __syncthreads();
    { // stage K (32x64, async->LDS) and V^T (64x32, VGPR transpose)
      int r = threadIdx.x >> 3, ci = (threadIdx.x & 7) * 8;
      int kp = kbase + r;
      uint4 zv = {};
      if (kp >= 0 && kp < S_LEN) {
        async_ld_b128(&Ks[r][ci], kb + (size_t)kp * D_MODEL + h * DHEAD + ci);
        zv = *(const uint4*)(vb + (size_t)kp * D_MODEL + h * DHEAD + ci);
      } else {
        uint4 zz = {};
        *(uint4*)&Ks[r][ci] = zz;   // zero-fill OOB keys (scores masked anyway)
      }
      const __bf16* pv = (const __bf16*)&zv;
#pragma unroll
      for (int e = 0; e < 8; ++e) Vt[ci + e][r] = pv[e];
    }
    wait_async0();
    __syncthreads();

    // scores for 32 keys (two 16-key tiles), K dim = 64 (2 wmma each)
    v8f s0 = {}, s1 = {};
    {
      v16bf b0a = fragA(&Ks[lr][0], lane, 0);
      v16bf b0b = fragA(&Ks[lr][0], lane, 32);
      s0 = wmma_bf16(qf0, b0a, s0);
      s0 = wmma_bf16(qf1, b0b, s0);
      v16bf b1a = fragA(&Ks[16 + lr][0], lane, 0);
      v16bf b1b = fragA(&Ks[16 + lr][0], lane, 32);
      s1 = wmma_bf16(qf0, b1a, s1);
      s1 = wmma_bf16(qf1, b1b, s1);
    }

    float sv0[8], sv1[8], scf[8];
#pragma unroll
    for (int j = 0; j < 8; ++j) {
      int qp = qrow0 + j + hi * 8;
      int kp0 = kbase + lr, kp1 = kbase + 16 + lr;
      bool v0 = (kp0 >= 0) && (kp0 < S_LEN) && (abs(kp0 - qp) <= WWIN);
      bool v1 = (kp1 >= 0) && (kp1 < S_LEN) && (abs(kp1 - qp) <= WWIN);
      sv0[j] = v0 ? s0[j] * 0.125f : -1.0e9f;
      sv1[j] = v1 ? s1[j] * 0.125f : -1.0e9f;
    }
#pragma unroll
    for (int j = 0; j < 8; ++j) {
      float rm = fmaxf(sv0[j], sv1[j]);
      for (int d = 1; d < 16; d <<= 1) rm = fmaxf(rm, __shfl_xor(rm, d, 16));
      float mnew = fmaxf(mrow[j], rm);
      float corr = __expf(mrow[j] - mnew);
      float p0 = __expf(sv0[j] - mnew);
      float p1 = __expf(sv1[j] - mnew);
      float rs = p0 + p1;
      for (int d = 1; d < 16; d <<= 1) rs += __shfl_xor(rs, d, 16);
      lrow[j] = lrow[j] * corr + rs;
      mrow[j] = mnew;
      scf[j] = corr;
      Pl[wid][j + hi * 8][lr]      = p0;
      Pl[wid][j + hi * 8][16 + lr] = p1;
    }
#pragma unroll
    for (int t = 0; t < 4; ++t)
#pragma unroll
      for (int j = 0; j < 8; ++j) ctxa[t][j] *= scf[j];

    // P (16x32) as A-fragment via per-wave LDS reshuffle (fp32 -> bf16)
    v16bf pf;
#pragma unroll
    for (int e = 0; e < 8; ++e) pf[e]     = (__bf16)Pl[wid][lr][hi * 8 + e];
#pragma unroll
    for (int e = 0; e < 8; ++e) pf[e + 8] = (__bf16)Pl[wid][lr][hi * 8 + 16 + e];

    // ctx += P @ V ; Vt rows are head-dims = Bt rows
#pragma unroll
    for (int t = 0; t < 4; ++t) {
      v16bf vf = fragA(&Vt[t * 16 + lr][0], lane, 0);
      ctxa[t] = wmma_bf16(pf, vf, ctxa[t]);
    }
  }

#pragma unroll
  for (int t = 0; t < 4; ++t)
#pragma unroll
    for (int j = 0; j < 8; ++j) {
      int qp = qrow0 + j + hi * 8;
      int d = h * DHEAD + t * 16 + lr;
      float val = ctxa[t][j] / fmaxf(lrow[j], 1e-20f);
      ctx[(size_t)qp * D_MODEL + d] = val;
      ctxb[(size_t)qp * D_MODEL + d] = (__bf16)val;
    }
}

// ---------------- elementwise / smalls ----------------
__global__ void embed_kernel(const int* __restrict__ ids,
                             const float* __restrict__ we,
                             const float* __restrict__ pe,
                             float* __restrict__ out) {
  int row = blockIdx.x;
  int id = ids[row];
  for (int i = threadIdx.x; i < D_MODEL; i += 256)
    out[(size_t)row * D_MODEL + i] =
        we[(size_t)id * D_MODEL + i] + pe[(size_t)row * D_MODEL + i];
}

__global__ void add_ln_kernel(const float* __restrict__ a,
                              const float* __restrict__ b,
                              const float* __restrict__ g,
                              const float* __restrict__ be,
                              float* __restrict__ out,
                              __bf16* __restrict__ outb) {
  __shared__ float red[256];
  int row = blockIdx.x;
  const float* ap = a + (size_t)row * D_MODEL;
  const float* bp = b ? b + (size_t)row * D_MODEL : nullptr;
  float vals[3];
  float s = 0.0f;
#pragma unroll
  for (int u = 0; u < 3; ++u) {
    int i = threadIdx.x + u * 256;
    float v = ap[i] + (bp ? bp[i] : 0.0f);
    vals[u] = v; s += v;
  }
  red[threadIdx.x] = s; __syncthreads();
  for (int st = 128; st > 0; st >>= 1) {
    if (threadIdx.x < st) red[threadIdx.x] += red[threadIdx.x + st];
    __syncthreads();
  }
  float mean = red[0] * (1.0f / D_MODEL);
  __syncthreads();
  float vs = 0.0f;
#pragma unroll
  for (int u = 0; u < 3; ++u) { float d = vals[u] - mean; vs += d * d; }
  red[threadIdx.x] = vs; __syncthreads();
  for (int st = 128; st > 0; st >>= 1) {
    if (threadIdx.x < st) red[threadIdx.x] += red[threadIdx.x + st];
    __syncthreads();
  }
  float rstd = rsqrtf(red[0] * (1.0f / D_MODEL) + 1e-5f);
#pragma unroll
  for (int u = 0; u < 3; ++u) {
    int i = threadIdx.x + u * 256;
    float o = (vals[u] - mean) * rstd * g[i] + be[i];
    out[(size_t)row * D_MODEL + i] = o;
    outb[(size_t)row * D_MODEL + i] = (__bf16)o;
  }
}

__global__ void accum_kernel(float* __restrict__ dst, const float* __restrict__ src,
                             int first, size_t n) {
  size_t i = (size_t)blockIdx.x * 256 + threadIdx.x;
  if (i < n) dst[i] = first ? src[i] : dst[i] + src[i];
}

__global__ void cast_bf16_kernel(const float* __restrict__ in, __bf16* __restrict__ out,
                                 size_t n) {
  size_t i = (size_t)blockIdx.x * 256 + threadIdx.x;
  if (i < n) out[i] = (__bf16)in[i];
}

__global__ void cast_transpose_kernel(const float* __restrict__ in,
                                      __bf16* __restrict__ out, int R, int C) {
  size_t i = (size_t)blockIdx.x * 256 + threadIdx.x;
  if (i < (size_t)R * C) {
    int r = (int)(i / C), c = (int)(i - (size_t)r * C);
    out[(size_t)c * R + r] = (__bf16)in[i];
  }
}

// ---------------- persistent LSTM (one block, sequential scan) ----------------
__global__ void lstm_kernel(const float* __restrict__ X4,
                            const float* __restrict__ Whh,
                            float* __restrict__ out, int reverse, int off) {
  __shared__ float hs[HID], cs[HID], gsh[H4];
  int tid = threadIdx.x;                     // blockDim = 512
  for (int i = tid; i < HID; i += 512) { hs[i] = 0.0f; cs[i] = 0.0f; }
  __syncthreads();
  for (int step = 0; step < S_LEN; ++step) {
    int t = reverse ? (S_LEN - 1 - step) : step;
    if (step + 1 < S_LEN) {                  // prefetch next timestep's gate row
      int tn = reverse ? (S_LEN - 2 - step) : (step + 1);
      __builtin_prefetch(X4 + (size_t)tn * H4 + tid * 3, 0, 0);
    }
    const float* xrow = X4 + (size_t)t * H4;
#pragma unroll
    for (int u = 0; u < 3; ++u) {
      int j = tid * 3 + u;
      float acc = xrow[j];
      const float* w = Whh + (size_t)j * HID;
      for (int kk = 0; kk < HID; ++kk) acc += hs[kk] * w[kk];
      gsh[j] = acc;
    }
    __syncthreads();
    if (tid < HID) {
      float ig = sigmoidf(gsh[tid]);
      float fg = sigmoidf(gsh[HID + tid]);
      float gg = tanhf(gsh[2 * HID + tid]);
      float og = sigmoidf(gsh[3 * HID + tid]);
      float cv = fg * cs[tid] + ig * gg;
      float hv = og * tanhf(cv);
      cs[tid] = cv; hs[tid] = hv;
      out[(size_t)t * D_MODEL + off + tid] = hv;
    }
    __syncthreads();
  }
}

// ---------------- event pooling / pair features / head+loss ----------------
__global__ void event_kernel(const int* __restrict__ label_event,
                             const float* __restrict__ tok2,
                             float* __restrict__ ev) {
  int n = blockIdx.x;
  int s = label_event[n * 3 + 0], e = label_event[n * 3 + 1];
  float inv = 1.0f / (float)(e - s);
  for (int d = threadIdx.x; d < D_MODEL; d += 256) {
    float acc = 0.0f;
    for (int t = s; t < e; ++t) acc += tok2[(size_t)t * D_MODEL + d];
    ev[(size_t)n * D_MODEL + d] = acc * inv;
  }
}

__global__ void pair_kernel(const int* __restrict__ pairs,
                            const float* __restrict__ ev,
                            float* __restrict__ feats,
                            __bf16* __restrict__ featsb) {
  int p = blockIdx.x;
  int i = pairs[p * 2], j = pairs[p * 2 + 1];
  size_t base = (size_t)p * (4 * D_MODEL);
  for (int d = threadIdx.x; d < D_MODEL; d += 256) {
    float a = ev[(size_t)i * D_MODEL + d];
    float b = ev[(size_t)j * D_MODEL + d];
    float f0 = a, f1 = b, f2 = a - b, f3 = a * b;
    feats[base + d] = f0;                 featsb[base + d] = (__bf16)f0;
    feats[base + D_MODEL + d] = f1;       featsb[base + D_MODEL + d] = (__bf16)f1;
    feats[base + 2 * D_MODEL + d] = f2;   featsb[base + 2 * D_MODEL + d] = (__bf16)f2;
    feats[base + 3 * D_MODEL + d] = f3;   featsb[base + 3 * D_MODEL + d] = (__bf16)f3;
  }
}

__global__ void head_loss_kernel(const float* __restrict__ h2o,
                                 const float* __restrict__ w3,
                                 const float* __restrict__ b3,
                                 const int* __restrict__ labels,
                                 float* __restrict__ d_out) {
  __shared__ float red[256];
  float local = 0.0f;
  for (int p = threadIdx.x; p < NPAIR; p += 256) {
    float l0 = b3[0], l1 = b3[1];
    const float* hp = h2o + (size_t)p * 256;
    for (int kk = 0; kk < 256; ++kk) {
      float hv = hp[kk];
      l0 += hv * w3[kk * 2 + 0];
      l1 += hv * w3[kk * 2 + 1];
    }
    d_out[1 + p * 2 + 0] = l0;
    d_out[1 + p * 2 + 1] = l1;
    float mx = fmaxf(l0, l1);
    float lse = mx + logf(__expf(l0 - mx) + __expf(l1 - mx));
    int lab = labels[p];
    float logp = (lab ? l1 : l0) - lse;
    float w = (lab == 0 ? 1.0f : 0.0f) + (float)lab;
    local += -logp * w;
  }
  red[threadIdx.x] = local; __syncthreads();
  for (int st = 128; st > 0; st >>= 1) {
    if (threadIdx.x < st) red[threadIdx.x] += red[threadIdx.x + st];
    __syncthreads();
  }
  if (threadIdx.x == 0) d_out[0] = red[0];
}

// ---------------- host orchestration ----------------
// d_in flattening (jax pytree, dicts sorted by key) — see round-0 notes.
extern "C" void kernel_launch(void* const* d_in, const int* in_sizes, int n_in,
                              void* d_out, int out_size, void* d_ws, size_t ws_size,
                              hipStream_t stream) {
  const int* ids     = (const int*)d_in[0];
  const int* levent  = (const int*)d_in[2];
  const int* pairs   = (const int*)d_in[3];
  const int* coref   = (const int*)d_in[4];
  const float* h1_b  = (const float*)d_in[5];
  const float* h1_w  = (const float*)d_in[6];
  const float* h2_b  = (const float*)d_in[7];
  const float* h2_w  = (const float*)d_in[8];
  const float* h3_b  = (const float*)d_in[9];
  const float* h3_w  = (const float*)d_in[10];
  const float* Lb1   = (const float*)d_in[11];
  const float* Lb2   = (const float*)d_in[12];
  const float* Lbk   = (const float*)d_in[13];
  const float* Lbo   = (const float*)d_in[14];
  const float* Lbq   = (const float*)d_in[15];
  const float* Lbv   = (const float*)d_in[16];
  const float* Lln1b = (const float*)d_in[17];
  const float* Lln1g = (const float*)d_in[18];
  const float* Lln2b = (const float*)d_in[19];
  const float* Lln2g = (const float*)d_in[20];
  const float* Lw1   = (const float*)d_in[21];
  const float* Lw2   = (const float*)d_in[22];
  const float* Lwk   = (const float*)d_in[23];
  const float* Lwo   = (const float*)d_in[24];
  const float* Lwq   = (const float*)d_in[25];
  const float* Lwv   = (const float*)d_in[26];
  const float* ln_eb = (const float*)d_in[27];
  const float* ln_eg = (const float*)d_in[28];
  const float* Whh_b = (const float*)d_in[29];
  const float* Whh_f = (const float*)d_in[30];
  const float* Wih_b = (const float*)d_in[31];
  const float* Wih_f = (const float*)d_in[32];
  const float* b_b   = (const float*)d_in[33];
  const float* b_f   = (const float*)d_in[34];
  const float* pose  = (const float*)d_in[35];
  const float* wemb  = (const float*)d_in[36];

  (void)in_sizes; (void)n_in; (void)out_size; (void)ws_size;

  uintptr_t cur = (uintptr_t)d_ws;
  auto alloc = [&](size_t bytes) -> void* {
    uintptr_t p = (cur + 255) & ~(uintptr_t)255;
    cur = p + bytes;
    return (void*)p;
  };
  const size_t SD = (size_t)S_LEN * D_MODEL;
  const size_t SF = (size_t)S_LEN * FFDIM;

  float* x     = (float*)alloc(SD * 4);
  float* tmpSD = (float*)alloc(SD * 4);
  float* qf    = (float*)alloc(SD * 4);
  float* kf    = (float*)alloc(SD * 4);
  float* vf    = (float*)alloc(SD * 4);
  float* ctxf  = (float*)alloc(SD * 4);
  float* ao    = (float*)alloc(SD * 4);
  float* ff1   = (float*)alloc(SF * 4);
  float* hsum  = (float*)alloc(SD * 4);
  float* X4f   = (float*)alloc((size_t)S_LEN * H4 * 4);
  float* X4b   = (float*)alloc((size_t)S_LEN * H4 * 4);
  float* tok2  = (float*)alloc(SD * 4);
  float* ev    = (float*)alloc((size_t)NEV * D_MODEL * 4);
  float* feats = (float*)alloc((size_t)NPAIR * 4 * D_MODEL * 4);
  float* h1o   = (float*)alloc((size_t)NPAIR * D_MODEL * 4);
  float* h2o   = (float*)alloc((size_t)NPAIR * 256 * 4);

  __bf16* xb     = (__bf16*)alloc(SD * 2);
  __bf16* qbb    = (__bf16*)alloc(SD * 2);
  __bf16* kbb    = (__bf16*)alloc(SD * 2);
  __bf16* vbb    = (__bf16*)alloc(SD * 2);
  __bf16* ctxbb  = (__bf16*)alloc(SD * 2);
  __bf16* ff1b   = (__bf16*)alloc(SF * 2);
  __bf16* tokb   = (__bf16*)alloc(SD * 2);
  __bf16* featsb = (__bf16*)alloc((size_t)NPAIR * 4 * D_MODEL * 2);
  __bf16* h1ob   = (__bf16*)alloc((size_t)NPAIR * D_MODEL * 2);
  __bf16* h2ob   = (__bf16*)alloc((size_t)NPAIR * 256 * 2);

  const size_t DD = (size_t)D_MODEL * D_MODEL;
  const size_t DF = (size_t)D_MODEL * FFDIM;
  __bf16* wqT = (__bf16*)alloc((size_t)NLAYER * DD * 2);
  __bf16* wkT = (__bf16*)alloc((size_t)NLAYER * DD * 2);
  __bf16* wvT = (__bf16*)alloc((size_t)NLAYER * DD * 2);
  __bf16* woT = (__bf16*)alloc((size_t)NLAYER * DD * 2);
  __bf16* w1T = (__bf16*)alloc((size_t)NLAYER * DF * 2);
  __bf16* w2T = (__bf16*)alloc((size_t)NLAYER * DF * 2);
  __bf16* WihFb = (__bf16*)alloc((size_t)H4 * D_MODEL * 2);
  __bf16* WihBb = (__bf16*)alloc((size_t)H4 * D_MODEL * 2);
  __bf16* h1wT  = (__bf16*)alloc((size_t)D_MODEL * (4 * D_MODEL) * 2);
  __bf16* h2wT  = (__bf16*)alloc((size_t)256 * D_MODEL * 2);

  dim3 blk(256);
  auto grid1 = [](size_t n) { return dim3((unsigned)((n + 255) / 256)); };

  for (int l = 0; l < NLAYER; ++l) {
    cast_transpose_kernel<<<grid1(DD), blk, 0, stream>>>(Lwq + l * DD, wqT + l * DD, D_MODEL, D_MODEL);
    cast_transpose_kernel<<<grid1(DD), blk, 0, stream>>>(Lwk + l * DD, wkT + l * DD, D_MODEL, D_MODEL);
    cast_transpose_kernel<<<grid1(DD), blk, 0, stream>>>(Lwv + l * DD, wvT + l * DD, D_MODEL, D_MODEL);
    cast_transpose_kernel<<<grid1(DD), blk, 0, stream>>>(Lwo + l * DD, woT + l * DD, D_MODEL, D_MODEL);
    cast_transpose_kernel<<<grid1(DF), blk, 0, stream>>>(Lw1 + l * DF, w1T + l * DF, D_MODEL, FFDIM);
    cast_transpose_kernel<<<grid1(DF), blk, 0, stream>>>(Lw2 + l * DF, w2T + l * DF, FFDIM, D_MODEL);
  }
  cast_bf16_kernel<<<grid1((size_t)H4 * D_MODEL), blk, 0, stream>>>(Wih_f, WihFb, (size_t)H4 * D_MODEL);
  cast_bf16_kernel<<<grid1((size_t)H4 * D_MODEL), blk, 0, stream>>>(Wih_b, WihBb, (size_t)H4 * D_MODEL);
  cast_transpose_kernel<<<grid1((size_t)4 * D_MODEL * D_MODEL), blk, 0, stream>>>(h1_w, h1wT, 4 * D_MODEL, D_MODEL);
  cast_transpose_kernel<<<grid1((size_t)D_MODEL * 256), blk, 0, stream>>>(h2_w, h2wT, D_MODEL, 256);

  embed_kernel<<<dim3(S_LEN), blk, 0, stream>>>(ids, wemb, pose, tmpSD);
  add_ln_kernel<<<dim3(S_LEN), blk, 0, stream>>>(tmpSD, nullptr, ln_eg, ln_eb, x, xb);

  auto gemm = [&](const __bf16* A, const __bf16* Bt, const float* bias, float* C,
                  __bf16* Cb, int M, int N, int K, int act) {
    dim3 g((unsigned)((M + 255) / 256), (unsigned)(N / 64));
    gemm_bf16_kernel<<<g, blk, 0, stream>>>(A, Bt, bias, C, Cb, M, N, K, act);
  };

  for (int l = 0; l < NLAYER; ++l) {
    gemm(xb, wqT + l * DD, Lbq + l * D_MODEL, qf, qbb, S_LEN, D_MODEL, D_MODEL, 0);
    gemm(xb, wkT + l * DD, Lbk + l * D_MODEL, kf, kbb, S_LEN, D_MODEL, D_MODEL, 0);
    gemm(xb, wvT + l * DD, Lbv + l * D_MODEL, vf, vbb, S_LEN, D_MODEL, D_MODEL, 0);
    attn_kernel<<<dim3(NHEAD, S_LEN / WWIN, 2), blk, 0, stream>>>(qbb, kbb, vbb, ctxf, ctxbb);
    gemm(ctxbb, woT + l * DD, Lbo + l * D_MODEL, ao, nullptr, S_LEN, D_MODEL, D_MODEL, 0);
    add_ln_kernel<<<dim3(S_LEN), blk, 0, stream>>>(x, ao, Lln1g + l * D_MODEL,
                                                   Lln1b + l * D_MODEL, x, xb);
    gemm(xb, w1T + l * DF, Lb1 + l * FFDIM, ff1, ff1b, S_LEN, FFDIM, D_MODEL, 1);
    gemm(ff1b, w2T + l * DF, Lb2 + l * D_MODEL, tmpSD, nullptr, S_LEN, D_MODEL, FFDIM, 0);
    add_ln_kernel<<<dim3(S_LEN), blk, 0, stream>>>(x, tmpSD, Lln2g + l * D_MODEL,
                                                   Lln2b + l * D_MODEL, x, xb);
    if (l >= NLAYER - 4)
      accum_kernel<<<grid1(SD), blk, 0, stream>>>(hsum, x, l == NLAYER - 4 ? 1 : 0, SD);
  }

  cast_bf16_kernel<<<grid1(SD), blk, 0, stream>>>(hsum, tokb, SD);
  gemm(tokb, WihFb, b_f, X4f, nullptr, S_LEN, H4, D_MODEL, 0);
  gemm(tokb, WihBb, b_b, X4b, nullptr, S_LEN, H4, D_MODEL, 0);
  lstm_kernel<<<dim3(1), dim3(512), 0, stream>>>(X4f, Whh_f, tok2, 0, 0);
  lstm_kernel<<<dim3(1), dim3(512), 0, stream>>>(X4b, Whh_b, tok2, 1, HID);

  event_kernel<<<dim3(NEV), blk, 0, stream>>>(levent, tok2, ev);
  pair_kernel<<<dim3(NPAIR), blk, 0, stream>>>(pairs, ev, feats, featsb);
  gemm(featsb, h1wT, h1_b, h1o, h1ob, NPAIR, D_MODEL, 4 * D_MODEL, 2);
  gemm(h1ob, h2wT, h2_b, h2o, h2ob, NPAIR, 256, D_MODEL, 2);
  head_loss_kernel<<<dim3(1), blk, 0, stream>>>(h2o, h3_w, h3_b, coref, (float*)d_out);
}
